// HierarchicalLSTMDecoder_16887811408414
// MI455X (gfx1250) — matrix operand order repaired and checked by reading, not compiled, not run on hardware
//
#include <hip/hip_runtime.h>
#include <hip/hip_bf16.h>

// ---------------------------------------------------------------------------
// HierarchicalLSTMDecoder for MI455X (gfx1250, wave32, WMMA)
//
// Strategy:
//  * kernel 1: context = concat(env,hist,goal) @ W_proj^T + b  (fp32, runs once)
//  * kernel 2: repack all recurrent weights to bf16 (W_ih0 K padded 258->288)
//  * kernel 3: persistent per-16-row workgroup; h/c state lives in LDS for all
//    60 steps; every gate GEMM uses v_wmma_f32_16x16x32_bf16 (f32 accum).
// ---------------------------------------------------------------------------

typedef __attribute__((ext_vector_type(16))) __bf16   v16bf;
typedef __attribute__((ext_vector_type(8)))  float    v8f;
typedef __attribute__((ext_vector_type(4)))  unsigned v4u;

union Frag { v16bf v; v4u q[2]; };

static constexpr int BB   = 2048;   // batch
static constexpr int HH   = 256;    // hidden
static constexpr int TT   = 60;     // steps
static constexpr int XKP  = 288;    // padded K for [pos(2)|ctx(256)] (mult of 32)
static constexpr int XS   = 296;    // LDS stride (halves) for x tile
static constexpr int HS   = 264;    // LDS stride (halves) for h tiles
static constexpr int GS   = 1032;   // LDS stride (floats) for gates

// ---------------------------------------------------------------------------
// WMMA helpers
// ---------------------------------------------------------------------------
__device__ __forceinline__ v8f wmma_bf16(const Frag& a, const Frag& b, v8f c) {
    return __builtin_amdgcn_wmma_f32_16x16x32_bf16(
        /*neg_a=*/false, a.v, /*neg_b=*/false, b.v,
        /*c_mod=*/(short)0, c, /*reuse_a=*/false, /*reuse_b=*/false);
}

// A fragment: 16x32 bf16 tile, row-major source with stride `s` halves.
// ISA layout: lane m=lane&15, half h=lane>>4; elems 0..7 = K=kb+8h+j,
// elems 8..15 = K=kb+16+8h+j  -> two contiguous 16B chunks.
__device__ __forceinline__ void load_a(Frag& f, const __bf16* base, int s,
                                       int kb, int lane) {
    const int m = lane & 15, h = lane >> 4;
    const __bf16* p = base + m * s + kb + 8 * h;
    f.q[0] = *(const v4u*)p;
    f.q[1] = *(const v4u*)(p + 16);
}

// B fragment: 32x16 tile of W^T, i.e. W row n (row-major NxK, stride Kw),
// lane n=n0+(lane&15), elems 0..7 = K=kb+8h+j, elems 8..15 = K=kb+16+8h+j.
__device__ __forceinline__ void load_b(Frag& f, const __bf16* W, int Kw,
                                       int n0, int kb, int lane) {
    const int n = n0 + (lane & 15), h = lane >> 4;
    const __bf16* p = W + (size_t)n * Kw + kb + 8 * h;
    f.q[0] = *(const v4u*)p;
    f.q[1] = *(const v4u*)(p + 16);
}

__device__ __forceinline__ float sigm(float x) { return 1.f / (1.f + __expf(-x)); }
__device__ __forceinline__ float tanh_fast(float x) {
    float e = __expf(2.f * x);
    return 1.f - 2.f / (e + 1.f);          // safe at +-inf
}

// gates(16x1024) = A1(16xK1)@W1^T + A2(16x256)@W2^T  ; wave handles 8 N-tiles
template <int K1ITERS>
__device__ __forceinline__ void gemm_two(float (*gates)[GS],
                                         const __bf16* a1, int a1s,
                                         const __bf16* __restrict__ W1, int W1k,
                                         const __bf16* a2, int a2s,
                                         const __bf16* __restrict__ W2,
                                         int wid, int lane) {
    for (int i = 0; i < 8; ++i) {
        const int n0 = (wid + 8 * i) * 16;
        v8f acc = {};
#pragma unroll
        for (int kk = 0; kk < K1ITERS; ++kk) {
            Frag fa, fb;
            load_a(fa, a1, a1s, 32 * kk, lane);
            load_b(fb, W1, W1k, n0, 32 * kk, lane);
            acc = wmma_bf16(fa, fb, acc);
        }
#pragma unroll
        for (int kk = 0; kk < 8; ++kk) {
            Frag fa, fb;
            load_a(fa, a2, a2s, 32 * kk, lane);
            load_b(fb, W2, 256, n0, 32 * kk, lane);
            acc = wmma_bf16(fa, fb, acc);
        }
        // C layout: VGPR r -> row r + 8*(lane>>4), col = n0 + (lane&15)
        const int n = lane & 15, hh = lane >> 4;
#pragma unroll
        for (int r = 0; r < 8; ++r) gates[hh * 8 + r][n0 + n] = acc[r];
    }
}

__device__ __forceinline__ void lstm_elem(float (*gates)[GS],
                                          const float* __restrict__ bsum,
                                          float (*c)[256], __bf16 (*hb)[HS],
                                          int tid) {
    for (int i = tid; i < 16 * 256; i += 256) {
        const int m = i >> 8, j = i & 255;
        const float iv = sigm(gates[m][j]           + bsum[j]);
        const float fv = sigm(gates[m][256 + j]     + bsum[256 + j]);
        const float gv = tanh_fast(gates[m][512 + j] + bsum[512 + j]);
        const float ov = sigm(gates[m][768 + j]     + bsum[768 + j]);
        const float cn = fv * c[m][j] + iv * gv;
        c[m][j] = cn;
        hb[m][j] = (__bf16)(ov * tanh_fast(cn));
    }
}

// ---------------------------------------------------------------------------
// Kernel 1: context projection (fp32, runs once; ~0.15% of total FLOPs)
// ---------------------------------------------------------------------------
__global__ void ctx_proj_kernel(const float* __restrict__ env,
                                const float* __restrict__ hist,
                                const float* __restrict__ goal,
                                const float* __restrict__ Wp,
                                const float* __restrict__ bp,
                                float* __restrict__ ctx) {
    const int idx = blockIdx.x * blockDim.x + threadIdx.x;
    const int b = idx >> 8, h = idx & 255;
    if (b >= BB) return;
    const float* w = Wp + h * 386;
    const float* e = env + b * 256;
    const float* hi = hist + b * 128;
    float s = bp[h];
    for (int k = 0; k < 256; ++k) s += w[k] * e[k];
    for (int k = 0; k < 128; ++k) s += w[256 + k] * hi[k];
    s += w[384] * goal[b * 2] + w[385] * goal[b * 2 + 1];
    ctx[idx] = s;
}

// ---------------------------------------------------------------------------
// Kernel 2: repack fp32 weights -> bf16 (W_ih0 padded K 258 -> 288 with zeros)
// ---------------------------------------------------------------------------
__global__ void repack_kernel(const float* __restrict__ Wih0,
                              const float* __restrict__ Whh0,
                              const float* __restrict__ Wih1,
                              const float* __restrict__ Whh1,
                              const float* __restrict__ Wout1,
                              __bf16* wih0p, __bf16* whh0, __bf16* wih1,
                              __bf16* whh1, __bf16* wout1) {
    const int idx = blockIdx.x * 256 + threadIdx.x;
    const int N0 = 1024 * XKP, N1 = 1024 * 256;
    if (idx < N0) {
        const int n = idx / XKP, k = idx % XKP;
        const float v = (k < 258) ? Wih0[n * 258 + k] : 0.f;
        wih0p[idx] = (__bf16)v;
    } else if (idx < N0 + N1) {
        const int j = idx - N0;           whh0[j] = (__bf16)Whh0[j];
    } else if (idx < N0 + 2 * N1) {
        const int j = idx - N0 - N1;      wih1[j] = (__bf16)Wih1[j];
    } else if (idx < N0 + 3 * N1) {
        const int j = idx - N0 - 2 * N1;  whh1[j] = (__bf16)Whh1[j];
    } else if (idx < N0 + 3 * N1 + 64 * 256) {
        const int j = idx - N0 - 3 * N1;  wout1[j] = (__bf16)Wout1[j];
    }
}

// ---------------------------------------------------------------------------
// Kernel 3: persistent decoder. 1 WG = 16 batch rows, 8 waves, 60 steps.
// ---------------------------------------------------------------------------
__global__ __launch_bounds__(256)
void lstm_decoder_kernel(const float* __restrict__ cur_pos,
                         const float* __restrict__ bih0, const float* __restrict__ bhh0,
                         const float* __restrict__ bih1, const float* __restrict__ bhh1,
                         const float* __restrict__ bo1g, const float* __restrict__ wo2g,
                         const float* __restrict__ bo2g,
                         const float* __restrict__ ctx,
                         const __bf16* __restrict__ Wih0p, const __bf16* __restrict__ Whh0,
                         const __bf16* __restrict__ Wih1,  const __bf16* __restrict__ Whh1,
                         const __bf16* __restrict__ Wout1,
                         float* __restrict__ out) {
    __shared__ alignas(16) __bf16 xb[16][XS];    // [pos|ctx] bf16, K padded
    __shared__ alignas(16) __bf16 h1b[16][HS];
    __shared__ alignas(16) __bf16 h2b[16][HS];
    __shared__ float c1[16][256];
    __shared__ float c2[16][256];
    __shared__ float gates[16][GS];
    __shared__ float mlp[16][68];
    __shared__ float bsum0[1024], bsum1[1024];
    __shared__ float bo1[64], wo2[128], bo2[2];

    const int tid  = threadIdx.x;
    const int wid  = tid >> 5;
    const int lane = tid & 31;
    const int m0   = blockIdx.x * 16;

    // ---- init ----
    for (int i = tid; i < 16 * XS; i += 256) (&xb[0][0])[i] = (__bf16)0.f;
    for (int i = tid; i < 1024; i += 256) {
        bsum0[i] = bih0[i] + bhh0[i];
        bsum1[i] = bih1[i] + bhh1[i];
    }
    if (tid < 64)  bo1[tid] = bo1g[tid];
    if (tid < 128) wo2[tid] = wo2g[tid];
    if (tid < 2)   bo2[tid] = bo2g[tid];
    __syncthreads();
    for (int i = tid; i < 16 * 256; i += 256) {
        const int m = i >> 8, j = i & 255;
        const float cv = ctx[(size_t)(m0 + m) * 256 + j];
        const __bf16 cb = (__bf16)cv;
        xb[m][2 + j] = cb;      // context portion of lstm0 input (constant)
        h1b[m][j] = cb;         // h0 init = context (both layers)
        h2b[m][j] = cb;
        c1[m][j] = 0.f;
        c2[m][j] = 0.f;
    }
    if (tid < 32) {
        const int m = tid >> 1, j = tid & 1;
        xb[m][j] = (__bf16)cur_pos[(m0 + m) * 2 + j];
    }
    __syncthreads();

    // ---- 60 decode steps ----
    for (int step = 0; step < TT; ++step) {
        // layer 0: gates = [pos|ctx] @ Wih0^T + h1 @ Whh0^T
        gemm_two<9>(gates, &xb[0][0], XS, Wih0p, XKP, &h1b[0][0], HS, Whh0,
                    wid, lane);
        __syncthreads();
        lstm_elem(gates, bsum0, c1, h1b, tid);
        __syncthreads();

        // layer 1: gates = h1 @ Wih1^T + h2 @ Whh1^T
        gemm_two<8>(gates, &h1b[0][0], HS, Wih1, 256, &h2b[0][0], HS, Whh1,
                    wid, lane);
        __syncthreads();
        lstm_elem(gates, bsum1, c2, h2b, tid);
        __syncthreads();

        // output head: relu(h2 @ Wout1^T + b1)  (N=64 -> 4 tiles, waves 0..3)
        if (wid < 4) {
            const int n0 = wid * 16;
            v8f acc = {};
#pragma unroll
            for (int kk = 0; kk < 8; ++kk) {
                Frag fa, fb;
                load_a(fa, &h2b[0][0], HS, 32 * kk, lane);
                load_b(fb, Wout1, 256, n0, 32 * kk, lane);
                acc = wmma_bf16(fa, fb, acc);
            }
            const int n = lane & 15, hh = lane >> 4;
#pragma unroll
            for (int r = 0; r < 8; ++r) {
                const float v = acc[r] + bo1[n0 + n];
                mlp[hh * 8 + r][n0 + n] = v > 0.f ? v : 0.f;
            }
        }
        __syncthreads();

        // pred = mlp @ Wout2^T + b2  (16 rows x 2 cols = 32 dots of 64)
        if (tid < 32) {
            const int m = tid >> 1, j = tid & 1;
            float s = bo2[j];
#pragma unroll 8
            for (int k = 0; k < 64; ++k) s += mlp[m][k] * wo2[j * 64 + k];
            out[((size_t)(m0 + m) * TT + step) * 2 + j] = s;  // (B,T,2) f32
            xb[m][j] = (__bf16)s;   // feed back as next step's pos
        }
        __syncthreads();
    }
}

// ---------------------------------------------------------------------------
extern "C" void kernel_launch(void* const* d_in, const int* in_sizes, int n_in,
                              void* d_out, int out_size, void* d_ws, size_t ws_size,
                              hipStream_t stream) {
    const float* env   = (const float*)d_in[0];
    const float* hist  = (const float*)d_in[1];
    const float* goal  = (const float*)d_in[2];
    const float* cpos  = (const float*)d_in[3];
    const float* Wproj = (const float*)d_in[4];
    const float* bproj = (const float*)d_in[5];
    const float* Wih0  = (const float*)d_in[6];
    const float* Whh0  = (const float*)d_in[7];
    const float* bih0  = (const float*)d_in[8];
    const float* bhh0  = (const float*)d_in[9];
    const float* Wih1  = (const float*)d_in[10];
    const float* Whh1  = (const float*)d_in[11];
    const float* bih1  = (const float*)d_in[12];
    const float* bhh1  = (const float*)d_in[13];
    const float* Wout1 = (const float*)d_in[14];
    const float* bout1 = (const float*)d_in[15];
    const float* Wout2 = (const float*)d_in[16];
    const float* bout2 = (const float*)d_in[17];
    float* out = (float*)d_out;

    // workspace layout (all 16B-aligned offsets)
    char* ws = (char*)d_ws;
    float*  ctx    = (float*)ws;                      // 2048*256*4 = 2,097,152
    __bf16* wih0p  = (__bf16*)(ws + 2097152);         // 1024*288*2 -> 2,686,976
    __bf16* whh0   = (__bf16*)(ws + 2686976);         // 1024*256*2 -> 3,211,264
    __bf16* wih1   = (__bf16*)(ws + 3211264);         //            -> 3,735,552
    __bf16* whh1   = (__bf16*)(ws + 3735552);         //            -> 4,259,840
    __bf16* wout1  = (__bf16*)(ws + 4259840);         // 64*256*2   -> 4,292,608

    // 1) context projection (fp32)
    ctx_proj_kernel<<<(BB * HH) / 256, 256, 0, stream>>>(env, hist, goal,
                                                         Wproj, bproj, ctx);
    // 2) bf16 weight repack
    {
        const int total = 1024 * XKP + 3 * 1024 * 256 + 64 * 256;
        repack_kernel<<<(total + 255) / 256, 256, 0, stream>>>(
            Wih0, Whh0, Wih1, Whh1, Wout1, wih0p, whh0, wih1, whh1, wout1);
    }
    // 3) persistent WMMA decoder: 128 WGs x 16 rows
    lstm_decoder_kernel<<<BB / 16, 256, 0, stream>>>(
        cpos, bih0, bhh0, bih1, bhh1, bout1, Wout2, bout2, ctx,
        wih0p, whh0, wih1, whh1, wout1, out);
}
